// GPTBlock_369367188059
// MI455X (gfx1250) — matrix operand order repaired and checked by reading, not compiled, not run on hardware
//
#include <hip/hip_runtime.h>
#include <hip/hip_bf16.h>

// ---------------------------------------------------------------------------
// GPT block (pre-norm) for MI455X / gfx1250, wave32 + WMMA bf16.
//   z1 = z + W_O( attn( rmsnorm(z,g1) ) );  out = z1 + W_proj( gelu( W_fc( rmsnorm(z1,g2) ) ) )
// All matmuls via V_WMMA_F32_16X16X32_BF16 (f32 accumulate).
// Compute-bound (~172 GFLOP vs ~4us of HBM traffic at 23.3 TB/s; all weights
// L2-resident in 192 MB), so GEMM waves run 32x64 tiles = 8 WMMAs / K-step.
// ---------------------------------------------------------------------------

typedef __bf16 bf16_t;
typedef __attribute__((ext_vector_type(16))) __bf16 v16bf;
typedef __attribute__((ext_vector_type(8)))  float  v8f;

constexpr int kCtx   = 2048;
constexpr int kDm    = 1024;
constexpr int kDff   = 4096;
constexpr int kHeads = 16;
constexpr int kDh    = 64;
constexpr int kBatch = 2;
constexpr int kM     = kBatch * kCtx;   // 4096 rows

// ------------------------------ helpers ------------------------------------

__device__ inline v8f zero_v8f() {
  v8f z;
#pragma unroll
  for (int i = 0; i < 8; ++i) z[i] = 0.0f;
  return z;
}

// Load 8 bf16 from p0 into lanes 0..7 and 8 bf16 from p1 into 8..15 (16B loads).
__device__ inline v16bf load_pair8(const bf16_t* p0, const bf16_t* p1) {
  union U { uint4 q; bf16_t h[8]; };
  U a, b;
  a.q = *reinterpret_cast<const uint4*>(p0);
  b.q = *reinterpret_cast<const uint4*>(p1);
  v16bf r;
#pragma unroll
  for (int i = 0; i < 8; ++i) { r[i] = a.h[i]; r[i + 8] = b.h[i]; }
  return r;
}

// A-fragment (16x32, row-major source [M,K]): lane m = lane&15, kbase=(lane>>4)*8,
// elements e<8 -> k=kbase+e ; e>=8 -> k=kbase+16+(e-8).
__device__ inline v16bf load_a_frag(const bf16_t* A, int row0, int K, int k0, int lane) {
  const bf16_t* p = A + (size_t)(row0 + (lane & 15)) * K + k0 + ((lane >> 4) * 8);
  return load_pair8(p, p + 16);
}

// B-fragment (32x16, from transposed weights [N,K]): lane n = lane&15, kbase=(lane>>4)*8,
// elements e<8 -> k=kbase+e ; e>=8 -> k=kbase+16+(e-8).
__device__ inline v16bf load_b_frag(const bf16_t* Bt, int col0, int K, int k0, int lane) {
  const bf16_t* p = Bt + (size_t)(col0 + (lane & 15)) * K + k0 + ((lane >> 4) * 8);
  return load_pair8(p, p + 16);
}

__device__ inline v8f wmma_bf16(v16bf a, v16bf b, v8f c) {
  return __builtin_amdgcn_wmma_f32_16x16x32_bf16(false, a, false, b, (short)0, c,
                                                 false, false);
}

__device__ inline float red16_max(float v) {
#pragma unroll
  for (int m = 1; m <= 8; m <<= 1) v = fmaxf(v, __shfl_xor(v, m, 32));
  return v;
}
__device__ inline float red16_sum(float v) {
#pragma unroll
  for (int m = 1; m <= 8; m <<= 1) v += __shfl_xor(v, m, 32);
  return v;
}

// ------------------------- weight cast + transpose -------------------------
// W [K,N] f32 row-major  ->  Wt [N,K] bf16 row-major (alpha folded in).
__global__ void cast_transpose_kernel(const float* __restrict__ W, bf16_t* __restrict__ Wt,
                                      int K, int N, float alpha) {
  int n = blockIdx.x * 16 + threadIdx.x;
  int k = blockIdx.y * 16 + threadIdx.y;
  Wt[(size_t)n * K + k] = (bf16_t)(W[(size_t)k * N + n] * alpha);
}

// ------------------------------ RMSNorm ------------------------------------
// One 256-thread block per row of 1024; float4 loads, LDS tree reduce, bf16 out.
__global__ __launch_bounds__(256) void rmsnorm_cast_kernel(const float* __restrict__ x,
                                                           const float* __restrict__ g,
                                                           bf16_t* __restrict__ y) {
  int row = blockIdx.x;
  int tid = threadIdx.x;
  const float4* xr = reinterpret_cast<const float4*>(x + (size_t)row * kDm);
  float4 v = xr[tid];
  float ss = v.x * v.x + v.y * v.y + v.z * v.z + v.w * v.w;
  __shared__ float red[256];
  red[tid] = ss;
  __syncthreads();
#pragma unroll
  for (int s = 128; s > 0; s >>= 1) {
    if (tid < s) red[tid] += red[tid + s];
    __syncthreads();
  }
  float inv = rsqrtf(red[0] * (1.0f / kDm) + 1e-5f);
  int c = tid * 4;
  bf16_t* yr = y + (size_t)row * kDm + c;
  yr[0] = (bf16_t)(v.x * inv * g[c + 0]);
  yr[1] = (bf16_t)(v.y * inv * g[c + 1]);
  yr[2] = (bf16_t)(v.z * inv * g[c + 2]);
  yr[3] = (bf16_t)(v.w * inv * g[c + 3]);
}

// ------------------------------- GEMM --------------------------------------
// C[M,N] = A[M,K](bf16) x Bt[N,K](bf16).
// 8 waves/block; each wave owns a 32x64 tile (2 A-frags x 4 B-frags ->
// 8 v_wmma per K=32 step); block tile 128x128. Epilogues:
//   0: f32 store     1: bf16 store        2: f32 store + residual
//   3: gelu -> bf16  4: bf16 store into V^T layout [B, Dm, Ctx]
template <int EPI>
__global__ __launch_bounds__(256) void gemm_bf16_kernel(
    const bf16_t* __restrict__ A, const bf16_t* __restrict__ Bt,
    const float* __restrict__ resid, float* __restrict__ of32,
    bf16_t* __restrict__ obf, int M, int N, int K) {
  int lane = threadIdx.x & 31;
  int w    = threadIdx.x >> 5;
  int base_m = blockIdx.x * 128 + (w & 3) * 32;   // 4 wave-rows of 32
  int base_n = blockIdx.y * 128 + (w >> 2) * 64;  // 2 wave-cols of 64

  v8f acc[2][4];
#pragma unroll
  for (int i = 0; i < 2; ++i)
#pragma unroll
    for (int j = 0; j < 4; ++j) acc[i][j] = zero_v8f();

  for (int k0 = 0; k0 < K; k0 += 32) {
    v16bf a0 = load_a_frag(A, base_m, K, k0, lane);
    v16bf a1 = load_a_frag(A, base_m + 16, K, k0, lane);
    v16bf b[4];
#pragma unroll
    for (int j = 0; j < 4; ++j) b[j] = load_b_frag(Bt, base_n + j * 16, K, k0, lane);
#pragma unroll
    for (int j = 0; j < 4; ++j) {
      acc[0][j] = wmma_bf16(a0, b[j], acc[0][j]);
      acc[1][j] = wmma_bf16(a1, b[j], acc[1][j]);
    }
    if (k0 + 32 < K) {  // L2-targeted prefetch of the next K-slab
      __builtin_prefetch(A + (size_t)(base_m + (lane & 15)) * K + k0 + 32);
      __builtin_prefetch(Bt + (size_t)(base_n + (lane & 15)) * K + k0 + 32);
      __builtin_prefetch(Bt + (size_t)(base_n + 32 + (lane & 15)) * K + k0 + 32);
    }
  }

  int hi = lane >> 4, lo = lane & 15;
#pragma unroll
  for (int i = 0; i < 2; ++i) {
#pragma unroll
    for (int j = 0; j < 4; ++j) {
#pragma unroll
      for (int r = 0; r < 8; ++r) {
        int row = base_m + i * 16 + r + hi * 8;
        int col = base_n + j * 16 + lo;
        float vv = acc[i][j][r];
        if (EPI == 0) {
          of32[(size_t)row * N + col] = vv;
        } else if (EPI == 1) {
          obf[(size_t)row * N + col] = (bf16_t)vv;
        } else if (EPI == 2) {
          of32[(size_t)row * N + col] = vv + resid[(size_t)row * N + col];
        } else if (EPI == 3) {
          float gl = 0.5f * vv * (1.0f + erff(vv * 0.70710678f));
          obf[(size_t)row * N + col] = (bf16_t)gl;
        } else {  // EPI == 4 : V transposed -> Vt[(b*Dm + col)*Ctx + t]
          int b = row >> 11;          // row / kCtx
          int t = row & (kCtx - 1);   // row % kCtx
          obf[((size_t)(b * kDm + col)) * kCtx + t] = (bf16_t)vv;
        }
      }
    }
  }
}

// ------------------------------ Attention ----------------------------------
// One wave per (b, h, 16-row Q block); online-softmax over 32-column blocks.
// Q/K bf16 [B*Ctx, Dm]; Vt bf16 [B, Dm, Ctx]; ctx bf16 [B*Ctx, Dm].
__global__ __launch_bounds__(128) void attn_kernel(const bf16_t* __restrict__ Qb,
                                                   const bf16_t* __restrict__ Kb,
                                                   const bf16_t* __restrict__ Vt,
                                                   bf16_t* __restrict__ ctx) {
  __shared__ bf16_t psh[4][16 * 32];  // per-wave P tile staging (C->A layout)
  int lane = threadIdx.x & 31;
  int w    = threadIdx.x >> 5;
  int task = blockIdx.x * 4 + w;            // B*H*(Ctx/16) = 4096 tasks
  int qblk = task & 127;
  int h    = (task >> 7) & 15;
  int b    = task >> 11;
  int q0   = qblk * 16;
  int hi = lane >> 4, lo = lane & 15;

  // Q A-fragments (1/sqrt(dh) already folded into W_Q).
  const bf16_t* qrow = Qb + (size_t)(b * kCtx + q0 + lo) * kDm + h * kDh + hi * 8;
  v16bf qa0 = load_pair8(qrow, qrow + 16);        // k = 0..31
  v16bf qa1 = load_pair8(qrow + 32, qrow + 48);   // k = 32..63

  v8f o[4];
  float mrow[8], lrow[8];
#pragma unroll
  for (int f = 0; f < 4; ++f) o[f] = zero_v8f();
#pragma unroll
  for (int r = 0; r < 8; ++r) { mrow[r] = -1e30f; lrow[r] = 0.0f; }

  for (int s0 = 0; s0 < q0 + 16; s0 += 32) {
    // ---- scores S = Q K^T for 32 columns (two 16x16 C fragments)
    v8f sf[2];
#pragma unroll
    for (int cc = 0; cc < 2; ++cc) {
      const bf16_t* kp =
          Kb + (size_t)(b * kCtx + s0 + cc * 16 + lo) * kDm + h * kDh + hi * 8;
      v16bf kb0 = load_pair8(kp, kp + 16);        // k = 0..31
      v16bf kb1 = load_pair8(kp + 32, kp + 48);   // k = 32..63
      v8f a = zero_v8f();
      a = wmma_bf16(qa0, kb0, a);
      a = wmma_bf16(qa1, kb1, a);
      sf[cc] = a;
    }

    // ---- causal mask on partial blocks
    if (s0 + 31 > q0 + 15) {
#pragma unroll
      for (int cc = 0; cc < 2; ++cc)
#pragma unroll
        for (int r = 0; r < 8; ++r) {
          int col = s0 + cc * 16 + lo;
          int row = q0 + r + hi * 8;
          if (col > row) sf[cc][r] = -1e30f;
        }
    }

    // ---- online softmax update; write P tile (bf16) into LDS
#pragma unroll
    for (int r = 0; r < 8; ++r) {
      float bm = red16_max(fmaxf(sf[0][r], sf[1][r]));
      float nm = fmaxf(mrow[r], bm);
      float sc = __expf(mrow[r] - nm);
      mrow[r] = nm;
      lrow[r] *= sc;
#pragma unroll
      for (int f = 0; f < 4; ++f) o[f][r] *= sc;
      float rs = 0.0f;
#pragma unroll
      for (int cc = 0; cc < 2; ++cc) {
        float p = __expf(sf[cc][r] - nm);
        psh[w][(r + hi * 8) * 32 + cc * 16 + lo] = (bf16_t)p;
        rs += p;
      }
      lrow[r] += red16_sum(rs);
    }
    asm volatile("s_wait_dscnt 0" ::: "memory");  // wave-local LDS RAW fence

    // ---- P as A-fragment from LDS (16x32)
    v16bf pa;
    {
      const bf16_t* pb = &psh[w][lo * 32 + hi * 8];
#pragma unroll
      for (int e = 0; e < 8; ++e) { pa[e] = pb[e]; pa[e + 8] = pb[e + 16]; }
    }

    // ---- O += P * V  (V^T layout makes B-fragments contiguous)
#pragma unroll
    for (int f = 0; f < 4; ++f) {
      const bf16_t* vp =
          Vt + ((size_t)(b * kDm + h * kDh + f * 16 + lo)) * kCtx + s0 + hi * 8;
      v16bf vb = load_pair8(vp, vp + 16);
      o[f] = wmma_bf16(pa, vb, o[f]);
    }
  }

  // ---- normalize and store ctx as [B*Ctx, Dm] (heads interleaved in columns)
#pragma unroll
  for (int r = 0; r < 8; ++r) {
    float inv = 1.0f / lrow[r];
#pragma unroll
    for (int f = 0; f < 4; ++f) {
      ctx[(size_t)(b * kCtx + q0 + r + hi * 8) * kDm + h * kDh + f * 16 + lo] =
          (bf16_t)(o[f][r] * inv);
    }
  }
}

// ------------------------------ launcher -----------------------------------

extern "C" void kernel_launch(void* const* d_in, const int* in_sizes, int n_in,
                              void* d_out, int out_size, void* d_ws, size_t ws_size,
                              hipStream_t stream) {
  (void)in_sizes; (void)n_in; (void)out_size; (void)ws_size;
  const float* z      = (const float*)d_in[0];
  const float* W_Q    = (const float*)d_in[1];
  const float* W_K    = (const float*)d_in[2];
  const float* W_V    = (const float*)d_in[3];
  const float* W_O    = (const float*)d_in[4];
  const float* W_fc   = (const float*)d_in[5];
  const float* W_proj = (const float*)d_in[6];
  const float* g1     = (const float*)d_in[7];
  const float* g2     = (const float*)d_in[8];
  float* out = (float*)d_out;

  // ---- workspace bump allocator (~120 MB total)
  char* p = (char*)d_ws;
  auto alloc = [&](size_t bytes) -> void* {
    void* r = (void*)p;
    p += (bytes + 255) & ~(size_t)255;
    return r;
  };
  bf16_t* WtQ  = (bf16_t*)alloc((size_t)kDm * kDm * 2);
  bf16_t* WtK  = (bf16_t*)alloc((size_t)kDm * kDm * 2);
  bf16_t* WtV  = (bf16_t*)alloc((size_t)kDm * kDm * 2);
  bf16_t* WtO  = (bf16_t*)alloc((size_t)kDm * kDm * 2);
  bf16_t* Wtfc = (bf16_t*)alloc((size_t)kDm * kDff * 2);
  bf16_t* Wtpj = (bf16_t*)alloc((size_t)kDff * kDm * 2);
  bf16_t* zn   = (bf16_t*)alloc((size_t)kM * kDm * 2);
  bf16_t* Qb   = (bf16_t*)alloc((size_t)kM * kDm * 2);
  bf16_t* Kb   = (bf16_t*)alloc((size_t)kM * kDm * 2);
  bf16_t* VtT  = (bf16_t*)alloc((size_t)kM * kDm * 2);
  bf16_t* ctxb = (bf16_t*)alloc((size_t)kM * kDm * 2);
  float*  z1   = (float*) alloc((size_t)kM * kDm * 4);
  bf16_t* hb   = (bf16_t*)alloc((size_t)kM * kDm * 2);
  bf16_t* fcb  = (bf16_t*)alloc((size_t)kM * kDff * 2);

  dim3 t16(16, 16);
  const float qscale = 0.125f;  // 1/sqrt(dh=64), folded into W_Q
  cast_transpose_kernel<<<dim3(kDm / 16, kDm / 16), t16, 0, stream>>>(W_Q, WtQ, kDm, kDm, qscale);
  cast_transpose_kernel<<<dim3(kDm / 16, kDm / 16), t16, 0, stream>>>(W_K, WtK, kDm, kDm, 1.0f);
  cast_transpose_kernel<<<dim3(kDm / 16, kDm / 16), t16, 0, stream>>>(W_V, WtV, kDm, kDm, 1.0f);
  cast_transpose_kernel<<<dim3(kDm / 16, kDm / 16), t16, 0, stream>>>(W_O, WtO, kDm, kDm, 1.0f);
  cast_transpose_kernel<<<dim3(kDff / 16, kDm / 16), t16, 0, stream>>>(W_fc, Wtfc, kDm, kDff, 1.0f);
  cast_transpose_kernel<<<dim3(kDm / 16, kDff / 16), t16, 0, stream>>>(W_proj, Wtpj, kDff, kDm, 1.0f);

  // zn = rmsnorm(z, g1)
  rmsnorm_cast_kernel<<<kM, 256, 0, stream>>>(z, g1, zn);

  // Q/K/V projections (V stored transposed for the P*V WMMA)
  gemm_bf16_kernel<1><<<dim3(kM / 128, kDm / 128), 256, 0, stream>>>(
      zn, WtQ, nullptr, nullptr, Qb, kM, kDm, kDm);
  gemm_bf16_kernel<1><<<dim3(kM / 128, kDm / 128), 256, 0, stream>>>(
      zn, WtK, nullptr, nullptr, Kb, kM, kDm, kDm);
  gemm_bf16_kernel<4><<<dim3(kM / 128, kDm / 128), 256, 0, stream>>>(
      zn, WtV, nullptr, nullptr, VtT, kM, kDm, kDm);

  // attention: B*H*(Ctx/16) = 4096 wave-tasks, 4 waves per block
  attn_kernel<<<kBatch * kHeads * (kCtx / 16) / 4, 128, 0, stream>>>(Qb, Kb, VtT, ctxb);

  // z1 = z + ctx @ W_O
  gemm_bf16_kernel<2><<<dim3(kM / 128, kDm / 128), 256, 0, stream>>>(
      ctxb, WtO, z, z1, nullptr, kM, kDm, kDm);

  // h = rmsnorm(z1, g2)
  rmsnorm_cast_kernel<<<kM, 256, 0, stream>>>(z1, g2, hb);

  // fc = gelu(h @ W_fc)
  gemm_bf16_kernel<3><<<dim3(kM / 128, kDff / 128), 256, 0, stream>>>(
      hb, Wtfc, nullptr, nullptr, fcb, kM, kDff, kDm);

  // out = z1 + fc @ W_proj
  gemm_bf16_kernel<2><<<dim3(kM / 128, kDm / 128), 256, 0, stream>>>(
      fcb, Wtpj, z1, out, nullptr, kM, kDm, kDff);
}
